// MyConvT_42039139893311
// MI455X (gfx1250) — matrix operand order repaired and checked by reading, not compiled, not run on hardware
//
#include <hip/hip_runtime.h>

typedef __attribute__((ext_vector_type(16))) __bf16          v16bf;
typedef __attribute__((ext_vector_type(8)))  float           v8f;
typedef __attribute__((ext_vector_type(8)))  unsigned short  us8;

typedef int v4i __attribute__((vector_size(16)));
typedef __attribute__((address_space(1))) v4i as1_v4i;
typedef __attribute__((address_space(3))) v4i as3_v4i;

#if __has_builtin(__builtin_amdgcn_global_load_async_to_lds_b128)
#define ASYNC_CP 1
#endif

#define CIN  192
#define COUT 128
#define Hh   64
#define Ww   64
#define OHh  128
#define OWw  128
#define MH   132
#define NB   8

// round-to-nearest-even f32 -> bf16 (bit pattern)
__device__ __forceinline__ unsigned short f2bf(float f) {
  unsigned int u = __float_as_uint(f);
  u += 0x7FFFu + ((u >> 16) & 1u);
  return (unsigned short)(u >> 16);
}

__device__ __forceinline__ void wait_async0() {
#ifdef ASYNC_CP
#if __has_builtin(__builtin_amdgcn_s_wait_asynccnt)
  __builtin_amdgcn_s_wait_asynccnt(0);
#else
  asm volatile("s_wait_asynccnt 0x0" ::: "memory");
#endif
#endif
}

// ---------------------------------------------------------------------------
// x [B,Cin,64,64] f32 (NCHW) -> xb bf16 NHWC [B,64,64,Cin], via padded LDS
// tile so both global reads and global writes are coalesced.
// One block per (b, ih) row.
// ---------------------------------------------------------------------------
__global__ void prep_x_kernel(const float* __restrict__ x,
                              unsigned short* __restrict__ xb) {
  __shared__ unsigned short t[64 * 194];  // pad 192->194: conflict-free
  const int b  = blockIdx.x >> 6;
  const int ih = blockIdx.x & 63;
  for (int i = threadIdx.x; i < 64 * CIN; i += 256) {
    int ci = i >> 6, iw = i & 63;  // consecutive i -> consecutive iw
    float f = x[((b * CIN + ci) * Hh + ih) * Ww + iw];
    t[iw * 194 + ci] = f2bf(f);
  }
  __syncthreads();
  unsigned short* dst = xb + (size_t)(b * Hh + ih) * Ww * CIN;
  for (int i = threadIdx.x; i < 64 * CIN; i += 256) {
    int iw = i / CIN, ci = i - iw * CIN;  // consecutive i -> contiguous NHWC
    dst[i] = t[iw * 194 + ci];
  }
}

// ---------------------------------------------------------------------------
// Repack weight [Cin,Cout,5,5] f32 -> bf16, per parity class p=(py,px):
//   wb[off_p + co*Kp + t*192 + ci], kh = py+2*th (dy = 1-th), kw = px+2*tw.
// ---------------------------------------------------------------------------
__global__ void prep_w_kernel(const float* __restrict__ w,
                              unsigned short* __restrict__ wb) {
  int i = blockIdx.x * 256 + threadIdx.x;
  if (i >= 614400) return;
  int p, base;
  if (i < 221184)      { p = 0; base = 0; }       // 3x3 taps
  else if (i < 368640) { p = 1; base = 221184; }  // 3x2
  else if (i < 516096) { p = 2; base = 368640; }  // 2x3
  else                 { p = 3; base = 516096; }  // 2x2
  int py = p >> 1, px = p & 1;
  int tw_n  = 3 - px;
  int ntaps = (3 - py) * tw_n;
  int local = i - base;
  int Kp = ntaps * CIN;
  int co  = local / Kp;
  int rem = local - co * Kp;
  int t   = rem / CIN;
  int ci  = rem - t * CIN;
  int th  = t / tw_n;
  int twi = t - th * tw_n;
  int kh = py + 2 * th;
  int kw = px + 2 * twi;
  wb[i] = f2bf(w[((ci * COUT + co) * 5 + kh) * 5 + kw]);
}

// 5x5 max-pool of the binary mask -> per-output-pixel gate
__global__ void prep_m_kernel(const int* __restrict__ mask,
                              float* __restrict__ pm) {
  int i = blockIdx.x * 256 + threadIdx.x;
  if (i >= NB * OHh * OWw) return;
  int b = i >> 14;
  int r = (i >> 7) & 127;
  int c = i & 127;
  const int* mp = mask + (b * MH + r) * MH + c;
  int any = 0;
#pragma unroll
  for (int dr = 0; dr < 5; ++dr)
#pragma unroll
    for (int dc = 0; dc < 5; ++dc)
      any |= mp[dr * MH + dc];
  pm[i] = any ? 1.0f : 0.0f;
}

// ---------------------------------------------------------------------------
// Stage one 6x18 halo x 32-channel chunk (bf16, NHWC source) into an LDS
// buffer with 40-element channel stride (80B cells: 16B-aligned, bank-clean).
// 432 aligned 16B segments; async copy when available, OOB cells zeroed.
// ---------------------------------------------------------------------------
__device__ __forceinline__ void stage_chunk(const unsigned short* __restrict__ xbb,
                                            unsigned short* buf, int ih0, int iw0,
                                            int ci0, int tid) {
  for (int s = tid; s < 432; s += 256) {
    int cell = s >> 2, sub = s & 3;
    int r = cell / 18, c = cell - r * 18;
    int ih = ih0 + r, iw = iw0 + c;
    unsigned short* lp = buf + cell * 40 + sub * 8;
    if ((unsigned)ih < 64u && (unsigned)iw < 64u) {
      const unsigned short* gp = xbb + ((ih * 64 + iw) * CIN + ci0 + sub * 8);
#ifdef ASYNC_CP
      __builtin_amdgcn_global_load_async_to_lds_b128((as1_v4i*)gp, (as3_v4i*)lp,
                                                     0, 0);
#else
      *(us8*)lp = *(const us8*)gp;
#endif
    } else {
      us8 z = {0, 0, 0, 0, 0, 0, 0, 0};
      *(us8*)lp = z;
    }
  }
}

// ---------------------------------------------------------------------------
// Main implicit-GEMM transposed conv, bf16 WMMA, f32 accumulate.
// 2048 blocks = 8 b * 4 parity * (16 u-blocks * 4 v-blocks); 8 wave32/block.
// Wave nw owns Cout [16nw,16nw+16); 4 accumulators cover 64 output pixels.
// Double-buffered async LDS staging; per-tap: 8 ds_load_b128 (A) issued
// up front, next-tap B preloaded, then 4 back-to-back WMMAs.
// ---------------------------------------------------------------------------
__global__ void __launch_bounds__(256) convt_wmma_kernel(
    const unsigned short* __restrict__ xb, const unsigned short* __restrict__ wb,
    const float* __restrict__ pm, const float* __restrict__ bias,
    float* __restrict__ out) {
  __shared__ unsigned short xs[2][6 * 18 * 40];

  const int bx   = blockIdx.x;
  const int tile = bx & 63;
  const int p    = (bx >> 6) & 3;
  const int b    = bx >> 8;
  const int py = p >> 1, px = p & 1;
  const int taps_h = 3 - py, taps_w = 3 - px;
  const int ntaps  = taps_h * taps_w;
  const int Kp = ntaps * CIN;
  const int offp[4] = {0, 221184, 368640, 516096};
  const unsigned short* wbp = wb + offp[p];

  const int u0 = (tile >> 2) * 4;    // parity-grid row base
  const int v0 = (tile & 3) * 16;    // parity-grid col base
  const int tid  = threadIdx.x;
  const int lane = tid & 31;
  const int nw   = tid >> 5;
  const int hi   = lane >> 4;
  const int nl   = lane & 15;

  v8f acc[4] = {};

  const int ih0 = u0 - 1, iw0 = v0 - 1;
  const int kb  = hi * 16;   // A per-lane K base (elements)
  const int klo = hi * 16;   // B per-lane K base (elements)
  const int coOff = (nw * 16 + nl) * Kp;

  const unsigned short* xbb = xb + (size_t)b * (Hh * Ww * CIN);

  stage_chunk(xbb, xs[0], ih0, iw0, 0, tid);
  wait_async0();
  __syncthreads();

  for (int c = 0; c < 6; ++c) {
    const int ci0 = c * 32;
    if (c + 1 < 6) stage_chunk(xbb, xs[(c + 1) & 1], ih0, iw0, ci0 + 32, tid);

    const unsigned short* cur = xs[c & 1];
    int th = 0, tw = 0;
    v16bf bcur = *(const v16bf*)(wbp + coOff + ci0 + klo);  // tap 0
    for (int t = 0; t < ntaps; ++t) {
      const int dy = 1 - th, dx = 1 - tw;
      union AV { us8 h[2]; v16bf v; } a[4];
#pragma unroll
      for (int mb = 0; mb < 4; ++mb) {
        const unsigned short* ap =
            cur + ((mb + dy + 1) * 18 + (nl + dx + 1)) * 40 + kb;
        a[mb].h[0] = *(const us8*)ap;
        a[mb].h[1] = *(const us8*)(ap + 8);
      }
      int tw2 = tw + 1, th2 = th;
      if (tw2 == taps_w) { tw2 = 0; th2 = th + 1; }
      v16bf bnext = bcur;
      if (t + 1 < ntaps)
        bnext = *(const v16bf*)(wbp + coOff + (th2 * taps_w + tw2) * CIN + ci0 + klo);
#pragma unroll
      for (int mb = 0; mb < 4; ++mb)
        acc[mb] = __builtin_amdgcn_wmma_f32_16x16x32_bf16(
            false, a[mb].v, false, bcur, (short)0, acc[mb], false, false);
      bcur = bnext; th = th2; tw = tw2;
    }

    wait_async0();
    __syncthreads();
  }

  // epilogue: gate by pooled mask, add bias
  const int co = nw * 16 + nl;
  const float bv = bias[co];
#pragma unroll
  for (int mb = 0; mb < 4; ++mb) {
    const int oh = 2 * (u0 + mb) + py;
#pragma unroll
    for (int r = 0; r < 8; ++r) {
      const int vl = r + 8 * hi;        // C row = pixel col index
      const int ow = 2 * (v0 + vl) + px;
      const float pmv = pm[(b * OHh + oh) * OWw + ow];
      const float val = pmv > 0.0f ? acc[mb][r] + bv : 0.0f;
      out[(((b * COUT + co) * OHh) + oh) * OWw + ow] = val;
    }
  }
}

extern "C" void kernel_launch(void* const* d_in, const int* in_sizes, int n_in,
                              void* d_out, int out_size, void* d_ws, size_t ws_size,
                              hipStream_t stream) {
  (void)in_sizes; (void)n_in; (void)out_size; (void)ws_size;
  const float* x      = (const float*)d_in[0];
  const int*   mask   = (const int*)d_in[1];
  const float* weight = (const float*)d_in[2];
  const float* bias   = (const float*)d_in[3];
  float* out = (float*)d_out;

  // workspace layout (256B-aligned sections)
  unsigned short* xb = (unsigned short*)d_ws;                          // 12,582,912 B
  unsigned short* wb = (unsigned short*)((char*)d_ws + 12582912);      //  1,228,800 B
  float*          pm = (float*)((char*)d_ws + 13811712);               //    524,288 B

  prep_x_kernel<<<NB * Hh, 256, 0, stream>>>(x, xb);
  prep_w_kernel<<<(614400 + 255) / 256, 256, 0, stream>>>(weight, wb);
  prep_m_kernel<<<(NB * OHh * OWw + 255) / 256, 256, 0, stream>>>(mask, pm);
  convt_wmma_kernel<<<NB * 4 * 64, 256, 0, stream>>>(xb, wb, pm, bias, out);
}
